// ImageTextMatching_19146964206107
// MI455X (gfx1250) — matrix operand order repaired and checked by reading, not compiled, not run on hardware
//
#include <hip/hip_runtime.h>
#include <hip/hip_bf16.h>

typedef __attribute__((ext_vector_type(4)))  float  v4f;
typedef __attribute__((ext_vector_type(8)))  float  v8f;
typedef __attribute__((ext_vector_type(4)))  __bf16 v4bf;
typedef __attribute__((ext_vector_type(8)))  __bf16 v8bf;
typedef __attribute__((ext_vector_type(16))) __bf16 v16bf;
typedef int v4i_g __attribute__((vector_size(16)));   // GCC-style vector: matches builtin proto

#define NUM_CLASSES 1000
#define B_SZ   16384
#define D_INN  1024
#define D_OUTN 256

// ---------------------------------------------------------------------------
// CDNA5 async global->LDS path (guarded; sync fallback keeps same semantics)
// ---------------------------------------------------------------------------
#if defined(__has_builtin)
#if __has_builtin(__builtin_amdgcn_global_load_async_to_lds_b128) && \
    __has_builtin(__builtin_amdgcn_s_wait_asynccnt)
#define USE_ASYNC_LDS 1
#endif
#endif

__device__ __forceinline__ void async_copy16(char* lds_dst, const __bf16* gsrc) {
#ifdef USE_ASYNC_LDS
    __builtin_amdgcn_global_load_async_to_lds_b128(
        (__attribute__((address_space(1))) v4i_g*)gsrc,
        (__attribute__((address_space(3))) v4i_g*)lds_dst, 0, 0);
#else
    *(v8bf*)lds_dst = *(const v8bf*)gsrc;   // global load + ds_store (sync)
#endif
}

#ifdef USE_ASYNC_LDS
#define ASYNC_WAIT(n) __builtin_amdgcn_s_wait_asynccnt(n)
#else
#define ASYNC_WAIT(n) ((void)0)
#endif

// ---------------------------------------------------------------------------
// Stage 1: bucketed segment sum (no float atomics)
// ---------------------------------------------------------------------------
__global__ __launch_bounds__(256) void zero_int_kernel(int* __restrict__ p, int n) {
    int i = blockIdx.x * 256 + threadIdx.x;
    if (i < n) p[i] = 0;
}

__global__ __launch_bounds__(256) void hist_kernel(const int* __restrict__ label,
                                                   int* __restrict__ cnti) {
    int i = blockIdx.x * 256 + threadIdx.x;
    if (i < B_SZ) atomicAdd(&cnti[label[i]], 1);
}

__global__ __launch_bounds__(1024) void scan_kernel(const int* __restrict__ cnti,
                                                    int* __restrict__ offs,
                                                    int* __restrict__ cursor) {
    __shared__ int tmp[1024];
    int t = threadIdx.x;
    int mine = cnti[t];
    tmp[t] = mine;
    __syncthreads();
    for (int off = 1; off < 1024; off <<= 1) {
        int add = (t >= off) ? tmp[t - off] : 0;
        __syncthreads();
        tmp[t] += add;
        __syncthreads();
    }
    int excl = tmp[t] - mine;
    offs[t] = excl;
    cursor[t] = excl;
}

__global__ __launch_bounds__(256) void fill_kernel(const int* __restrict__ label,
                                                   int* __restrict__ cursor,
                                                   int* __restrict__ rowidx) {
    int i = blockIdx.x * 256 + threadIdx.x;
    if (i < B_SZ) {
        int p = atomicAdd(&cursor[label[i]], 1);
        rowidx[p] = i;
    }
}

__global__ __launch_bounds__(256) void classsum_kernel(const float* __restrict__ x,
                                                       const int* __restrict__ rowidx,
                                                       const int* __restrict__ offs,
                                                       const int* __restrict__ cnti,
                                                       float* __restrict__ ls,
                                                       float* __restrict__ cntf) {
    __shared__ int rows[2048];
    int c = blockIdx.x;
    int start = offs[c], n = cnti[c];
    float acc[4] = {0.f, 0.f, 0.f, 0.f};
    for (int base = 0; base < n; base += 2048) {
        int m = min(n - base, 2048);
        __syncthreads();
        for (int j = threadIdx.x; j < m; j += 256) rows[j] = rowidx[start + base + j];
        __syncthreads();
        for (int j = 0; j < m; ++j) {
            const float* xr = x + (size_t)rows[j] * D_INN;
#pragma unroll
            for (int q = 0; q < 4; ++q) acc[q] += xr[threadIdx.x + 256 * q];
        }
    }
#pragma unroll
    for (int q = 0; q < 4; ++q)
        ls[(size_t)c * D_INN + threadIdx.x + 256 * q] = acc[q];
    if (threadIdx.x == 0) cntf[c] = (float)n;
}

__global__ __launch_bounds__(256) void total_kernel(const float* __restrict__ ls,
                                                    float* __restrict__ total) {
    int d = blockIdx.x * 256 + threadIdx.x;
    float s = 0.0f;
    for (int c = 0; c < NUM_CLASSES; ++c) s += ls[(size_t)c * D_INN + d];
    total[d] = s;
}

// ---------------------------------------------------------------------------
// Stage 2: split-precision (bf16 hi + bf16 residual) materialization
// ---------------------------------------------------------------------------
__global__ __launch_bounds__(256) void split4_kernel(const float* __restrict__ src,
                                                     __bf16* __restrict__ hi,
                                                     __bf16* __restrict__ lo, int n4) {
    int i = blockIdx.x * 256 + threadIdx.x;
    if (i >= n4) return;
    v4f v = ((const v4f*)src)[i];
    v4bf h, l;
#pragma unroll
    for (int j = 0; j < 4; ++j) {
        __bf16 hj = (__bf16)v[j];
        h[j] = hj;
        l[j] = (__bf16)(v[j] - (float)hj);
    }
    ((v4bf*)hi)[i] = h;
    ((v4bf*)lo)[i] = l;
}

__global__ __launch_bounds__(256) void mean_split_kernel(const float* __restrict__ ls,
                                                         const float* __restrict__ cntf,
                                                         const float* __restrict__ total,
                                                         __bf16* __restrict__ mh,
                                                         __bf16* __restrict__ ml) {
    int idx = blockIdx.x * 256 + threadIdx.x;  // covers 1024*1024
    int c = idx >> 10, d = idx & 1023;
    float m = 0.0f;
    if (c < NUM_CLASSES) {
        float oc = (float)B_SZ - cntf[c];
        if (oc > 0.0f) m = (total[d] - ls[idx]) / fmaxf(oc, 1.0f);
    }
    __bf16 h = (__bf16)m;
    mh[idx] = h;
    ml[idx] = (__bf16)(m - (float)h);
}

// ---------------------------------------------------------------------------
// Stage 3: bf16x3 WMMA GEMM with async-LDS-staged A tiles.
// Block = 128M x 64N, 8 waves (2 m-slots x 4 n-slots), wave = 64M x 16N.
// acc += Ah*Bh + Ah*Bl + Al*Bh  (drops only ~2^-16 Al*Bl term; f32 accumulate)
// ---------------------------------------------------------------------------
#define KCHUNK 32
#define NCHUNK (D_INN / KCHUNK)          // 32
#define PITCH  80                        // 64B data + 16B pad, 16B aligned
#define PLANE_BYTES (128 * PITCH)        // rows x pitch
#define BUF_BYTES (2 * PLANE_BYTES)      // hi + lo planes
#define SMEM_BYTES (3 * BUF_BYTES)       // triple buffer = 61440 B

__device__ __forceinline__ v16bf concat8(v8bf a, v8bf b) {
    return __builtin_shufflevector(a, b, 0, 1, 2, 3, 4, 5, 6, 7,
                                   8, 9, 10, 11, 12, 13, 14, 15);
}

// Stage one 128x32 hi+lo A chunk into LDS buffer bufIdx (4 async b128 per wave).
__device__ __forceinline__ void issue_chunk(const __bf16* __restrict__ Ah,
                                            const __bf16* __restrict__ Al,
                                            char* smem, int bufIdx, int blockM,
                                            int chunk, int w, int lane) {
    int l4 = lane & 3;       // 16B segment within a 64B row
    int lr = lane >> 2;      // row within group of 8
    int kc = chunk * KCHUNK;
#pragma unroll
    for (int o = 0; o < 4; ++o) {
        int oo = w * 4 + o;                  // 0..31
        int p = oo >> 4;                     // plane: 0=hi 1=lo
        int r = (oo & 15) * 8 + lr;          // 0..127
        const __bf16* src = (p ? Al : Ah) + (size_t)(blockM + r) * D_INN + kc + 8 * l4;
        char* dst = smem + (size_t)bufIdx * BUF_BYTES + (size_t)p * PLANE_BYTES
                    + r * PITCH + 16 * l4;
        async_copy16(dst, src);
    }
}

// Full K loop for one wave's 64x16 tile; A from LDS, B hi/lo from global.
__device__ __forceinline__ void gemm3_tile_lds(const __bf16* __restrict__ Ah,
                                               const __bf16* __restrict__ Al,
                                               const __bf16* __restrict__ Bh,
                                               const __bf16* __restrict__ Bl,
                                               char* smem, int blockM,
                                               int mslot, int n0w, v8f acc[4]) {
    int lane = threadIdx.x & 31;
    int w = threadIdx.x >> 5;
    int rl16 = lane & 15;
    int kh = lane >> 4;
    const __bf16* bh = Bh + (size_t)(n0w + rl16) * D_INN + 8 * kh;
    const __bf16* bl = Bl + (size_t)(n0w + rl16) * D_INN + 8 * kh;
    int arow = mslot * 64 + rl16;

    issue_chunk(Ah, Al, smem, 0, blockM, 0, w, lane);
    issue_chunk(Ah, Al, smem, 1, blockM, 1, w, lane);

    for (int ic = 0; ic < NCHUNK; ++ic) {
        if (ic + 1 < NCHUNK) { ASYNC_WAIT(4); } else { ASYNC_WAIT(0); }
        __syncthreads();                               // chunk ic visible block-wide
        if (ic + 2 < NCHUNK)                            // refill buffer freed at ic-1
            issue_chunk(Ah, Al, smem, (ic + 2) % 3, blockM, ic + 2, w, lane);

        int k = ic * KCHUNK;
        v16bf bhf = concat8(*(const v8bf*)(bh + k), *(const v8bf*)(bh + k + 16));
        v16bf blf = concat8(*(const v8bf*)(bl + k), *(const v8bf*)(bl + k + 16));
        const char* sb = smem + (size_t)(ic % 3) * BUF_BYTES;
#pragma unroll
        for (int s = 0; s < 4; ++s) {
            const char* ph = sb + (arow + s * 16) * PITCH + 16 * kh;
            const char* pl = ph + PLANE_BYTES;
            v16bf ahf = concat8(*(const v8bf*)ph, *(const v8bf*)(ph + 32));
            v16bf alf = concat8(*(const v8bf*)pl, *(const v8bf*)(pl + 32));
            acc[s] = __builtin_amdgcn_wmma_f32_16x16x32_bf16(false, ahf, false, bhf,
                                                             (short)0, acc[s], false, false);
            acc[s] = __builtin_amdgcn_wmma_f32_16x16x32_bf16(false, ahf, false, blf,
                                                             (short)0, acc[s], false, false);
            acc[s] = __builtin_amdgcn_wmma_f32_16x16x32_bf16(false, alf, false, bhf,
                                                             (short)0, acc[s], false, false);
        }
    }
}

// mproj[c][n] = mean[c] @ W2^T + (b1[n] + b2[n])
__global__ __launch_bounds__(256) void gemm_mean_kernel(const __bf16* __restrict__ mh,
                                                        const __bf16* __restrict__ ml,
                                                        const __bf16* __restrict__ w2h,
                                                        const __bf16* __restrict__ w2l,
                                                        const float* __restrict__ b1,
                                                        const float* __restrict__ b2,
                                                        float* __restrict__ mproj) {
    __shared__ __align__(16) char smem[SMEM_BYTES];
    int w = threadIdx.x >> 5;
    int mslot = w >> 2, nslot = w & 3;
    int blockM = blockIdx.y * 128;
    int n0w = blockIdx.x * 64 + nslot * 16;
    v8f acc[4] = {};
    gemm3_tile_lds(mh, ml, w2h, w2l, smem, blockM, mslot, n0w, acc);
    int lane = threadIdx.x & 31;
    int rl16 = lane & 15, kh = lane >> 4;
    int col = n0w + rl16;
    float bias = b1[col] + b2[col];
    int m0w = blockM + mslot * 64;
#pragma unroll
    for (int s = 0; s < 4; ++s)
#pragma unroll
        for (int v = 0; v < 8; ++v) {
            int row = m0w + s * 16 + kh * 8 + v;
            mproj[(size_t)row * D_OUTN + col] = acc[s][v] + bias;
        }
}

// out[i][n] = x[i] @ W1^T + mproj[label[i]][n]
__global__ __launch_bounds__(256) void gemm_x_kernel(const __bf16* __restrict__ xh,
                                                     const __bf16* __restrict__ xl,
                                                     const __bf16* __restrict__ w1h,
                                                     const __bf16* __restrict__ w1l,
                                                     const float* __restrict__ mproj,
                                                     const int* __restrict__ label,
                                                     float* __restrict__ out) {
    __shared__ __align__(16) char smem[SMEM_BYTES];
    int w = threadIdx.x >> 5;
    int mslot = w >> 2, nslot = w & 3;
    int blockM = blockIdx.y * 128;
    int n0w = blockIdx.x * 64 + nslot * 16;
    v8f acc[4] = {};
    gemm3_tile_lds(xh, xl, w1h, w1l, smem, blockM, mslot, n0w, acc);
    int lane = threadIdx.x & 31;
    int rl16 = lane & 15, kh = lane >> 4;
    int col = n0w + rl16;
    int m0w = blockM + mslot * 64;
#pragma unroll
    for (int s = 0; s < 4; ++s)
#pragma unroll
        for (int v = 0; v < 8; ++v) {
            int row = m0w + s * 16 + kh * 8 + v;
            out[(size_t)row * D_OUTN + col] = acc[s][v] + mproj[(size_t)label[row] * D_OUTN + col];
        }
}

// ---------------------------------------------------------------------------
extern "C" void kernel_launch(void* const* d_in, const int* in_sizes, int n_in,
                              void* d_out, int out_size, void* d_ws, size_t ws_size,
                              hipStream_t stream) {
    const float* x    = (const float*)d_in[0];
    const int*  label = (const int*)d_in[1];
    const float* W1w  = (const float*)d_in[2];
    const float* W1b  = (const float*)d_in[3];
    const float* W2w  = (const float*)d_in[4];
    const float* W2b  = (const float*)d_in[5];
    float* out = (float*)d_out;

    char* p = (char*)d_ws;
    auto alloc = [&](size_t bytes) {
        char* r = p;
        p += (bytes + 255) & ~(size_t)255;
        return r;
    };
    int*    cnti   = (int*)alloc(1024 * sizeof(int));
    int*    offs   = (int*)alloc(1024 * sizeof(int));
    int*    cursor = (int*)alloc(1024 * sizeof(int));
    int*    rowidx = (int*)alloc(B_SZ * sizeof(int));
    float*  ls     = (float*)alloc((size_t)NUM_CLASSES * D_INN * sizeof(float));
    float*  cntf   = (float*)alloc(1024 * sizeof(float));
    float*  total  = (float*)alloc(1024 * sizeof(float));
    float*  mproj  = (float*)alloc((size_t)1024 * D_OUTN * sizeof(float));
    __bf16* xh     = (__bf16*)alloc((size_t)B_SZ * D_INN * sizeof(__bf16));
    __bf16* xl     = (__bf16*)alloc((size_t)B_SZ * D_INN * sizeof(__bf16));
    __bf16* w1h    = (__bf16*)alloc((size_t)D_OUTN * D_INN * sizeof(__bf16));
    __bf16* w1l    = (__bf16*)alloc((size_t)D_OUTN * D_INN * sizeof(__bf16));
    __bf16* w2h    = (__bf16*)alloc((size_t)D_OUTN * D_INN * sizeof(__bf16));
    __bf16* w2l    = (__bf16*)alloc((size_t)D_OUTN * D_INN * sizeof(__bf16));
    __bf16* mh     = (__bf16*)alloc((size_t)1024 * D_INN * sizeof(__bf16));
    __bf16* ml     = (__bf16*)alloc((size_t)1024 * D_INN * sizeof(__bf16));

    // --- bucketed segment sum ---
    zero_int_kernel<<<4, 256, 0, stream>>>(cnti, 1024);
    hist_kernel<<<B_SZ / 256, 256, 0, stream>>>(label, cnti);
    scan_kernel<<<1, 1024, 0, stream>>>(cnti, offs, cursor);
    fill_kernel<<<B_SZ / 256, 256, 0, stream>>>(label, cursor, rowidx);
    classsum_kernel<<<NUM_CLASSES, 256, 0, stream>>>(x, rowidx, offs, cnti, ls, cntf);
    total_kernel<<<D_INN / 256, 256, 0, stream>>>(ls, total);

    // --- split-precision materialization ---
    mean_split_kernel<<<(1024 * D_INN) / 256, 256, 0, stream>>>(ls, cntf, total, mh, ml);
    split4_kernel<<<(B_SZ * D_INN / 4) / 256, 256, 0, stream>>>(x, xh, xl, B_SZ * D_INN / 4);
    split4_kernel<<<(D_OUTN * D_INN / 4) / 256, 256, 0, stream>>>(W1w, w1h, w1l, D_OUTN * D_INN / 4);
    split4_kernel<<<(D_OUTN * D_INN / 4) / 256, 256, 0, stream>>>(W2w, w2h, w2l, D_OUTN * D_INN / 4);

    // --- bf16x3 WMMA GEMMs (async-LDS staged A) ---
    gemm_mean_kernel<<<dim3(64 / 64, 1024 / 128), 256, 0, stream>>>(mh, ml, w2h, w2l,
                                                                    W1b, W2b, mproj);
    gemm_x_kernel<<<dim3(D_OUTN / 64, B_SZ / 128), 256, 0, stream>>>(xh, xl, w1h, w1l,
                                                                     mproj, label, out);
}
